// TorchMHACore_27522150432896
// MI455X (gfx1250) — compile-verified, matrix operand order checked by reading, not run on hardware
//
#include <hip/hip_runtime.h>
#include <hip/hip_fp16.h>
#include <hip/hip_bf16.h>

// ---------------------------------------------------------------------------
// Fused RoPE + causal GQA flash attention for gfx1250 (MI455X), wave32.
// B=2, S=1024, Hq=32, Hkv=8, D=128.  fp32 in/out, f16 WMMA compute.
// Async-to-LDS double-buffered K/V staging; V kept transposed in LDS so both
// QK^T and PV B-fragments are contiguous ds_load_2addr_b64 gathers.
// ---------------------------------------------------------------------------

typedef __attribute__((ext_vector_type(16))) _Float16 v16h;
typedef __attribute__((ext_vector_type(8)))  _Float16 v8h;
typedef __attribute__((ext_vector_type(8)))  float    v8f;

#define HQ     32
#define HKV    8
#define GQA    4
#define HDIM   128
#define SEQ    1024
#define NBATCH 2
#define QTILE  128         // q rows per workgroup
#define KTILE  32          // kv rows per inner iteration
#define NWAVES 8
#define LDQ    (HDIM + 4)  // padded LDS strides (halves)
#define LDK    (HDIM + 4)
#define LDVT   (KTILE + 8) // transposed V stride: 40 halves = 80 B (16B aligned)
#define LDP    (KTILE + 4)

// -ln(500000)/64
#define NEG_LN_THETA_OVER_64 (-0.20503832f)

// ---------------------------------------------------------------------------
// Q staging: QTILE x 128 fp32 from GLOBAL -> LDS f16 with RoPE.
// ---------------------------------------------------------------------------
__device__ __forceinline__ void stage_q_f16(const float* __restrict__ src,
                                            int srcStride, int posBase,
                                            _Float16* dst, int tid) {
  const int items = QTILE * 8;  // one item = 8 (d, d+64) pairs of one row
  for (int i = tid; i < items; i += 256) {
    const int row = i >> 3;
    const int grp = i & 7;
    const float* p = src + row * srcStride + grp * 8;
    const float4 a0 = *(const float4*)(p);
    const float4 a1 = *(const float4*)(p + 4);
    const float4 b0 = *(const float4*)(p + 64);
    const float4 b1 = *(const float4*)(p + 68);
    const float xa[8] = {a0.x, a0.y, a0.z, a0.w, a1.x, a1.y, a1.z, a1.w};
    const float xb[8] = {b0.x, b0.y, b0.z, b0.w, b1.x, b1.y, b1.z, b1.w};
    const float pos = (float)(posBase + row);
    _Float16* drow = dst + row * LDQ + grp * 8;
#pragma unroll
    for (int u = 0; u < 8; ++u) {
      const int d = grp * 8 + u;
      const float freq = __expf((float)d * NEG_LN_THETA_OVER_64);
      const float ang  = pos * freq;
      const float cs = __cosf(ang);
      const float sn = __sinf(ang);
      drow[u]      = (_Float16)(xa[u] * cs - xb[u] * sn);
      drow[64 + u] = (_Float16)(xa[u] * sn + xb[u] * cs);
    }
  }
}

// ---------------------------------------------------------------------------
// Async DMA one 32x128 fp32 tile: GLOBAL -> LDS raw buffer (ASYNCcnt-tracked).
// ---------------------------------------------------------------------------
__device__ __forceinline__ void async_load_tile(const float* __restrict__ src,
                                                int srcStrideFloats,
                                                float* rawDst, int tid) {
#pragma unroll
  for (int it = 0; it < 4; ++it) {
    const int i     = tid + it * 256;          // 0..1023
    const int row   = i >> 5;                  // 0..31
    const int chunk = i & 31;                  // 0..31 (16B chunks)
    const unsigned long long ga =
        (unsigned long long)(size_t)(src + (size_t)row * srcStrideFloats + chunk * 4);
    const unsigned lds = (unsigned)(size_t)(rawDst + row * HDIM + chunk * 4);
    asm volatile("global_load_async_to_lds_b128 %0, %1, off"
                 :
                 : "v"(lds), "v"(ga)
                 : "memory");
  }
}

__device__ __forceinline__ void wait_async_all() {
  asm volatile("s_wait_asynccnt 0x0" ::: "memory");
}

// ---------------------------------------------------------------------------
// K conversion: raw fp32 32x128 (LDS) -> f16 row-major tile (LDS) with RoPE.
// ---------------------------------------------------------------------------
__device__ __forceinline__ void convert_k_f16(const float* raw, int posBase,
                                              _Float16* dst, int tid) {
  const int row = tid >> 3;
  const int grp = tid & 7;
  const float* p = raw + row * HDIM + grp * 8;
  const float4 a0 = *(const float4*)(p);
  const float4 a1 = *(const float4*)(p + 4);
  const float4 b0 = *(const float4*)(p + 64);
  const float4 b1 = *(const float4*)(p + 68);
  const float xa[8] = {a0.x, a0.y, a0.z, a0.w, a1.x, a1.y, a1.z, a1.w};
  const float xb[8] = {b0.x, b0.y, b0.z, b0.w, b1.x, b1.y, b1.z, b1.w};
  const float pos = (float)(posBase + row);
  _Float16* drow = dst + row * LDK + grp * 8;
#pragma unroll
  for (int u = 0; u < 8; ++u) {
    const int d = grp * 8 + u;
    const float freq = __expf((float)d * NEG_LN_THETA_OVER_64);
    const float ang  = pos * freq;
    const float cs = __cosf(ang);
    const float sn = __sinf(ang);
    drow[u]      = (_Float16)(xa[u] * cs - xb[u] * sn);
    drow[64 + u] = (_Float16)(xa[u] * sn + xb[u] * cs);
  }
}

// ---------------------------------------------------------------------------
// V conversion with transpose: raw fp32 32x128 (LDS) -> f16 sVt[dim][kvrow].
// Thread handles dims {2p, 2p+1} x rows [g, g+8): contiguous b64 reads,
// two contiguous 16B stores.
// ---------------------------------------------------------------------------
__device__ __forceinline__ void convert_v_transposed_f16(const float* raw,
                                                         _Float16* dstT,
                                                         int tid) {
  const int p = tid >> 2;        // 0..63 -> dims 2p, 2p+1
  const int g = (tid & 3) * 8;   // first kv row of this patch
  v8h o0, o1;
#pragma unroll
  for (int r = 0; r < 8; ++r) {
    const float2 x = *(const float2*)(raw + (g + r) * HDIM + 2 * p);
    o0[r] = (_Float16)x.x;
    o1[r] = (_Float16)x.y;
  }
  *(v8h*)(dstT + (2 * p) * LDVT + g)     = o0;
  *(v8h*)(dstT + (2 * p + 1) * LDVT + g) = o1;
}

// k index inside a 32-wide chunk for 16-bit A/B fragments (ISA 7.12.2)
__device__ __forceinline__ int frag_k(int v, int h, int khi) {
  return ((v >> 2) << 4) + (khi << 3) + ((v & 3) << 1) + h;
}

// A fragment (16x32 f16): row = rowBase + (lane&15), k in AMD table order per lane.
__device__ __forceinline__ v16h frag_a(const _Float16* s, int ld, int rowBase,
                                       int kOff, int lane) {
  const int m   = rowBase + (lane & 15);
  const int khi = (lane >> 4) & 1;
  v16h a;
#pragma unroll
  for (int v = 0; v < 8; ++v)
#pragma unroll
    for (int h = 0; h < 2; ++h)
      a[v * 2 + h] = s[m * ld + kOff + frag_k(v, h, khi)];
  return a;
}

// B fragment from a row-major-in-n tile: B[k][n] = s[(nBase+n)*ld + kOff + k].
// Used for K^T (n = kv row) and transposed V (n = output dim).
__device__ __forceinline__ v16h frag_b_rowmaj(const _Float16* s, int ld,
                                              int nBase, int kOff, int lane) {
  const int n   = nBase + (lane & 15);
  const int khi = (lane >> 4) & 1;
  v16h b;
#pragma unroll
  for (int v = 0; v < 8; ++v)
#pragma unroll
    for (int h = 0; h < 2; ++h)
      b[v * 2 + h] = s[n * ld + kOff + frag_k(v, h, khi)];
  return b;
}

// Butterfly max over the 16-lane half via ds_swizzle (xor patterns).
template <int MASK>
__device__ __forceinline__ float swz_max_step(float x) {
  const int o =
      __builtin_amdgcn_ds_swizzle(__float_as_int(x), (MASK << 10) | 0x1f);
  return fmaxf(x, __int_as_float(o));
}
__device__ __forceinline__ float rowmax16(float x) {
  x = swz_max_step<1>(x);
  x = swz_max_step<2>(x);
  x = swz_max_step<4>(x);
  x = swz_max_step<8>(x);
  return x;
}

__global__ __launch_bounds__(256) void rope_flash_attn_kernel(
    const float* __restrict__ q, const float* __restrict__ k,
    const float* __restrict__ v, float* __restrict__ out) {
  // Overlay: Q f16 staging region is dead after fragment preload; reuse it as
  // the raw fp32 buffer pair for async K/V DMA (32 KB <= 33 KB).
  __shared__ __align__(16) unsigned char smemQ[QTILE * LDQ * 2];  // 33792 B
  _Float16* sQ   = (_Float16*)smemQ;
  float*    rawK = (float*)smemQ;                                  // 16 KB
  float*    rawV = (float*)(smemQ + KTILE * HDIM * sizeof(float)); // 16 KB
  __shared__ _Float16 sK[KTILE * LDK];        // 8448 B
  __shared__ _Float16 sVt[HDIM * LDVT];       // 10240 B (V transposed)
  __shared__ _Float16 sP[NWAVES * 16 * LDP];  // 9216 B

  const int tid  = threadIdx.x;
  const int lane = tid & 31;
  const int wave = tid >> 5;

  const int nQT = SEQ / QTILE;  // 8
  const int b   = blockIdx.x / (HQ * nQT);
  const int rem = blockIdx.x % (HQ * nQT);
  const int h   = rem / nQT;
  const int qb  = rem % nQT;
  const int hk  = h / GQA;

  const int qRowBase = qb * QTILE;
  const int qRow0    = qRowBase + wave * 16;

  // ---- stage Q (RoPE, fp32->f16) into LDS, preload A-fragments ----
  const float* qSrc = q + ((size_t)(b * SEQ + qRowBase) * HQ + h) * HDIM;
  stage_q_f16(qSrc, HQ * HDIM, qRowBase, sQ, tid);
  __syncthreads();

  v16h aQ[4];
#pragma unroll
  for (int kc = 0; kc < 4; ++kc)
    aQ[kc] = frag_a(sQ, LDQ, wave * 16, kc * 32, lane);
  __syncthreads();  // sQ region now dead -> safe for raw DMA overlay

  const float* kHead = k + ((size_t)b * SEQ * HKV + hk) * HDIM;
  const float* vHead = v + ((size_t)b * SEQ * HKV + hk) * HDIM;

  // kick off DMA of tile 0
  async_load_tile(kHead, HKV * HDIM, rawK, tid);
  async_load_tile(vHead, HKV * HDIM, rawV, tid);

  v8f   cO[8];
  float mRow[8], lRow[8], corr[8];
#pragma unroll
  for (int j = 0; j < 8; ++j)
    cO[j] = (v8f){0.f, 0.f, 0.f, 0.f, 0.f, 0.f, 0.f, 0.f};
#pragma unroll
  for (int r = 0; r < 8; ++r) { mRow[r] = -1e30f; lRow[r] = 0.f; }

  v16h onesB;
#pragma unroll
  for (int e = 0; e < 16; ++e) onesB[e] = (_Float16)1.0f;

  const float scale  = 0.08838834764831845f;  // 1/sqrt(128)
  const int   ktLast = qb * 4 + 3;            // causal bound (block uniform)
  const int   rowHi  = ((lane >> 4) & 1) * 8;
  _Float16*   sPw    = sP + wave * 16 * LDP;

  for (int kt = 0; kt <= ktLast; ++kt) {
    // our wave's DMA has landed; barrier publishes all waves' portions
    wait_async_all();
    __syncthreads();

    // raw fp32 (LDS) -> f16 tiles (LDS): RoPE K row-major, V transposed
    convert_k_f16(rawK, kt * KTILE, sK, tid);
    convert_v_transposed_f16(rawV, sVt, tid);
    __syncthreads();

    // overlap: DMA next tile while this one is consumed by WMMA
    if (kt < ktLast) {
      const float* nk = kHead + (size_t)((kt + 1) * KTILE) * HKV * HDIM;
      const float* nv = vHead + (size_t)((kt + 1) * KTILE) * HKV * HDIM;
      async_load_tile(nk, HKV * HDIM, rawK, tid);
      async_load_tile(nv, HKV * HDIM, rawV, tid);
    }

    // ---- S = Q @ K^T : 16x32 per wave, 8 WMMAs ----
    v8f c0 = (v8f){0.f, 0.f, 0.f, 0.f, 0.f, 0.f, 0.f, 0.f};
    v8f c1 = c0;
#pragma unroll
    for (int kc = 0; kc < 4; ++kc) {
      const v16h b0 = frag_b_rowmaj(sK, LDK, 0, kc * 32, lane);
      c0 = __builtin_amdgcn_wmma_f32_16x16x32_f16(false, aQ[kc], false, b0,
                                                  (short)0, c0, false, false);
      const v16h b1 = frag_b_rowmaj(sK, LDK, 16, kc * 32, lane);
      c1 = __builtin_amdgcn_wmma_f32_16x16x32_f16(false, aQ[kc], false, b1,
                                                  (short)0, c1, false, false);
    }

    // ---- online softmax: max via ds_swizzle butterfly, sum via WMMA ----
    const int n0 = kt * KTILE + (lane & 15);
    const int n1 = n0 + 16;
#pragma unroll
    for (int r = 0; r < 8; ++r) {
      const int qpos = qRow0 + r + rowHi;
      float s0 = c0[r] * scale;
      float s1 = c1[r] * scale;
      if (n0 > qpos) s0 = -1e30f;
      if (n1 > qpos) s1 = -1e30f;
      const float mx   = rowmax16(fmaxf(s0, s1));
      const float mNew = fmaxf(mRow[r], mx);
      corr[r] = __expf(mRow[r] - mNew);
      mRow[r] = mNew;
      const float p0 = __expf(s0 - mNew);
      const float p1 = __expf(s1 - mNew);
#pragma unroll
      for (int j = 0; j < 8; ++j) cO[j][r] *= corr[r];
      sPw[(r + rowHi) * LDP + (lane & 15)]      = (_Float16)p0;
      sPw[(r + rowHi) * LDP + 16 + (lane & 15)] = (_Float16)p1;
    }

    // P fragment feeds both the row-sum WMMA and the PV WMMAs
    const v16h aP = frag_a(sPw, LDP, 0, 0, lane);

    // row sums: P(16x32) @ ones(32x16) replicates sum_k P[m][k] in every col
    v8f cS = (v8f){0.f, 0.f, 0.f, 0.f, 0.f, 0.f, 0.f, 0.f};
    cS = __builtin_amdgcn_wmma_f32_16x16x32_f16(false, aP, false, onesB,
                                                (short)0, cS, false, false);
#pragma unroll
    for (int r = 0; r < 8; ++r) lRow[r] = lRow[r] * corr[r] + cS[r];

    // ---- O += P @ V : 8 WMMAs, B from transposed V (contiguous gathers) ----
#pragma unroll
    for (int j = 0; j < 8; ++j) {
      const v16h bv = frag_b_rowmaj(sVt, LDVT, j * 16, 0, lane);
      cO[j] = __builtin_amdgcn_wmma_f32_16x16x32_f16(false, aP, false, bv,
                                                     (short)0, cO[j], false,
                                                     false);
    }
  }

  // ---- normalize and store fp32 output: (B,1,S,HQ*D) ----
#pragma unroll
  for (int r = 0; r < 8; ++r) {
    const int   qpos = qRow0 + r + rowHi;
    const float inv  = 1.0f / lRow[r];
    float* orow = out + ((size_t)(b * SEQ + qpos) * HQ + h) * HDIM;
#pragma unroll
    for (int j = 0; j < 8; ++j)
      orow[j * 16 + (lane & 15)] = cO[j][r] * inv;
  }
}

extern "C" void kernel_launch(void* const* d_in, const int* in_sizes, int n_in,
                              void* d_out, int out_size, void* d_ws,
                              size_t ws_size, hipStream_t stream) {
  (void)in_sizes; (void)n_in; (void)d_ws; (void)ws_size; (void)out_size;
  const float* q = (const float*)d_in[0];
  const float* k = (const float*)d_in[1];
  const float* v = (const float*)d_in[2];
  float* out = (float*)d_out;
  const int blocks = NBATCH * HQ * (SEQ / QTILE);  // 512
  rope_flash_attn_kernel<<<blocks, 256, 0, stream>>>(q, k, v, out);
}
